// Attention_Fusion_Bimodal_LNSpecific_2_39092792328570
// MI455X (gfx1250) — matrix operand order repaired and checked by reading, not compiled, not run on hardware
//
#include <hip/hip_runtime.h>
#include <hip/hip_bf16.h>
#include <math.h>

// ---------------- problem constants ----------------
#define BD      4
#define CCH     768
#define HH      32
#define WW      32
#define HW      1024          // HH*WW
#define DMODEL  512
#define LQ      2048          // 2*HW
#define NLAYERS 6
#define NLEVELS 2
#define NHEADS  8
#define NPOINTS 4
#define FFN_DIM 1024
#define HD      64            // DMODEL/NHEADS

// ---------------- WMMA types ----------------
typedef __attribute__((ext_vector_type(16))) __bf16 v16bf;
typedef __attribute__((ext_vector_type(8)))  float  v8f;

union Frag { v16bf v; unsigned u[8]; };

__device__ __forceinline__ unsigned short f2bf(float f) {
  unsigned u = __float_as_uint(f);
  unsigned r = ((u >> 16) & 1u) + 0x7FFFu;   // round-to-nearest-even
  return (unsigned short)((u + r) >> 16);
}
__device__ __forceinline__ unsigned pack_bf(float lo, float hi) {
  return (unsigned)f2bf(lo) | ((unsigned)f2bf(hi) << 16);
}

// ---------------- tiled bf16 WMMA GEMM ----------------
// C[m,n] = residual? + (relu?)(sum_k A[m,k]*B[k,n] + bias)
// A element (m,k): A + m*a_rs + k*a_cs
// B element (k,n): B + k*b_rs + n*b_cs
// Requires: M%128==0, N%64==0, K%32==0  (true for every GEMM in this net)
#define TM 128
#define TN 64
#define TK 32
#define LDSR 42   // ushort row stride = 21 words: gcd(21,64)=1 -> no bank conflicts

__device__ __forceinline__ void load_a_frag(const unsigned short* As, int mbase,
                                            int lane, Frag& f) {
  int m = mbase + (lane & 15);
  int h = lane >> 4;
  const unsigned short* row = As + m * LDSR;
#pragma unroll
  for (int v = 0; v < 8; ++v) {
    int kb = ((v < 4) ? 0 : 16) + h * 8 + (v & 3) * 2;   // K pair base
    f.u[v] = *(const unsigned int*)(row + kb);
  }
}

__device__ __forceinline__ void load_b_frag(const unsigned short* Bs, int nbase,
                                            int lane, Frag& f) {
  int n = nbase + (lane & 15);
  int h = lane >> 4;
  const unsigned short* row = Bs + n * LDSR;               // Bs stored [n][k]
#pragma unroll
  for (int v = 0; v < 8; ++v) {
    int k = ((v < 4) ? 0 : 16) + h * 8 + (v & 3) * 2;
    f.u[v] = *(const unsigned int*)(row + k);
  }
}

// BIAS_MODE: 0 none, 1 per-n, 2 per-m
// A_CONTIG: a_cs == 1 (K contiguous in A).  B_KCONTIG: b_rs == 1 (K contiguous in B).
template <int BIAS_MODE, bool RELU, bool HAS_RES, bool A_CONTIG, bool B_KCONTIG>
__global__ __launch_bounds__(256)
void gemm_bf16_wmma(const float* __restrict__ A, long a_batch, int a_rs, int a_cs,
                    const float* __restrict__ B, long b_batch, int b_rs, int b_cs,
                    float* __restrict__ C, long c_batch, int ldc,
                    const float* __restrict__ bias,
                    const float* __restrict__ res, long res_batch, int res_ld,
                    int K) {
  __shared__ unsigned short As[TM * LDSR];
  __shared__ unsigned short Bs[TN * LDSR];
  unsigned* As32 = (unsigned*)As;   // row stride 21 words
  unsigned* Bs32 = (unsigned*)Bs;

  const int b  = blockIdx.z;
  const int m0 = blockIdx.y * TM;
  const int n0 = blockIdx.x * TN;
  A += (long)b * a_batch;
  B += (long)b * b_batch;
  C += (long)b * c_batch;
  if (HAS_RES) res += (long)b * res_batch;

  const int tid  = threadIdx.x;
  const int lane = tid & 31;
  const int wid  = tid >> 5;
  const int wm   = (wid >> 1) * 32;   // 0,32,64,96
  const int wn   = (wid & 1) * 32;    // 0,32

  // ---- per-thread staging coordinates (constant across the k-loop) ----
  const int apc = tid & 15;           // A pair-column; k = 2*apc (+k0)
  const int ar0 = tid >> 4;           // A row base; iter i adds 16
  const int bnn = tid & 63;           // B column (consecutive lanes -> coalesced)
  const int bk0 = (tid >> 6) * 2;     // B k base; iter i adds 8

  const float* Abase = A + (long)(m0 + ar0) * a_rs + (long)(2 * apc) * a_cs;
  const float* Bbase = B + (long)(n0 + bnn) * b_cs + (long)bk0 * b_rs;
  unsigned* AsW = As32 + ar0 * 21 + apc;
  unsigned* BsW = Bs32 + bnn * 21 + (bk0 >> 1);

  float2 ra[8];
  float2 rb[4];

  auto fetch = [&](int k0) {
#pragma unroll
    for (int i = 0; i < 8; ++i) {
      if (A_CONTIG) {
        ra[i] = *(const float2*)(Abase + (long)i * 16 * a_rs + k0);
      } else {
        const float* p = Abase + (long)i * 16 * a_rs + (long)k0 * a_cs;
        ra[i].x = p[0]; ra[i].y = p[a_cs];
      }
    }
#pragma unroll
    for (int i = 0; i < 4; ++i) {
      if (B_KCONTIG) {
        rb[i] = *(const float2*)(Bbase + i * 8 + k0);
      } else {
        const float* p = Bbase + (long)(i * 8 + k0) * b_rs;
        rb[i].x = p[0]; rb[i].y = p[b_rs];
      }
    }
  };
  auto stage = [&]() {
#pragma unroll
    for (int i = 0; i < 8; ++i) AsW[i * 16 * 21] = pack_bf(ra[i].x, ra[i].y);
#pragma unroll
    for (int i = 0; i < 4; ++i) BsW[i * 4] = pack_bf(rb[i].x, rb[i].y);
  };

  v8f zero = {0.f, 0.f, 0.f, 0.f, 0.f, 0.f, 0.f, 0.f};
  v8f acc[2][2];
  acc[0][0] = zero; acc[0][1] = zero; acc[1][0] = zero; acc[1][1] = zero;

  fetch(0);
  for (int k0 = 0;;) {
    stage();                     // waits on in-flight loads here, then LDS stores
    __syncthreads();
    const int kn = k0 + TK;
    const bool more = (kn < K);
    if (more) fetch(kn);         // prefetch next tile: overlaps frag loads + WMMA

    Frag af[2], bf2[2];
    load_a_frag(As, wm + 0,  lane, af[0]);
    load_a_frag(As, wm + 16, lane, af[1]);
    load_b_frag(Bs, wn + 0,  lane, bf2[0]);
    load_b_frag(Bs, wn + 16, lane, bf2[1]);

#pragma unroll
    for (int i = 0; i < 2; ++i)
#pragma unroll
      for (int j = 0; j < 2; ++j)
        acc[i][j] = __builtin_amdgcn_wmma_f32_16x16x32_bf16(
            false, af[i].v, false, bf2[j].v, (short)0, acc[i][j], false, false);

    if (!more) break;
    __syncthreads();
    k0 = kn;
  }

  // ---- epilogue: straight-line, no runtime branches ----
  const int h  = lane >> 4;
  const int nn = lane & 15;
#pragma unroll
  for (int i = 0; i < 2; ++i) {
#pragma unroll
    for (int j = 0; j < 2; ++j) {
      int gcol = n0 + wn + j * 16 + nn;
      float bn = (BIAS_MODE == 1) ? bias[gcol] : 0.f;
#pragma unroll
      for (int r = 0; r < 8; ++r) {
        int grow = m0 + wm + i * 16 + r + 8 * h;
        float v = acc[i][j][r] + bn;
        if (BIAS_MODE == 2) v += bias[grow];
        if (RELU) v = fmaxf(v, 0.f);
        if (HAS_RES) v += res[(long)grow * res_ld + gcol];
        C[(long)grow * ldc + gcol] = v;
      }
    }
  }
}

// ---------------- group norm (strided; serves tokens and NCHW) ----------------
__global__ __launch_bounds__(256)
void group_norm_kernel(float* __restrict__ x, long batch_stride,
                       const float* __restrict__ g, const float* __restrict__ bta,
                       int cpg, int np, int ch_stride, int p_stride) {
  int grp = blockIdx.x;
  int b   = blockIdx.y;
  x += (long)b * batch_stride;
  int n = cpg * np;
  float s = 0.f, s2 = 0.f;
  for (int i = threadIdx.x; i < n; i += blockDim.x) {
    int j = i % cpg, p = i / cpg;
    float v = x[(long)p * p_stride + (long)(grp * cpg + j) * ch_stride];
    s += v; s2 += v * v;
  }
  __shared__ float rs[256], rq[256];
  rs[threadIdx.x] = s; rq[threadIdx.x] = s2;
  __syncthreads();
  for (int off = 128; off > 0; off >>= 1) {
    if ((int)threadIdx.x < off) { rs[threadIdx.x] += rs[threadIdx.x + off];
                                  rq[threadIdx.x] += rq[threadIdx.x + off]; }
    __syncthreads();
  }
  float mean = rs[0] / n;
  float var  = rq[0] / n - mean * mean;
  float inv  = rsqrtf(var + 1e-5f);
  for (int i = threadIdx.x; i < n; i += blockDim.x) {
    int j = i % cpg, p = i / cpg;
    int ch = grp * cpg + j;
    long idx = (long)p * p_stride + (long)ch * ch_stride;
    x[idx] = (x[idx] - mean) * inv * g[ch] + bta[ch];
  }
}

// ---------------- modality-specific layer norm (in place) ----------------
__global__ __launch_bounds__(128)
void layer_norm_kernel(float* __restrict__ src, const float* __restrict__ g,
                       const float* __restrict__ bta) {
  int q = blockIdx.x;
  int b = blockIdx.y;
  int mod = (q >= HW) ? 1 : 0;
  float* x = src + ((long)b * LQ + q) * DMODEL;
  const float* gg = g + mod * DMODEL;
  const float* bb = bta + mod * DMODEL;
  float s = 0.f, s2 = 0.f;
  for (int i = threadIdx.x; i < DMODEL; i += 128) { float v = x[i]; s += v; s2 += v * v; }
  __shared__ float rs[128], rq[128];
  rs[threadIdx.x] = s; rq[threadIdx.x] = s2;
  __syncthreads();
  for (int off = 64; off > 0; off >>= 1) {
    if ((int)threadIdx.x < off) { rs[threadIdx.x] += rs[threadIdx.x + off];
                                  rq[threadIdx.x] += rq[threadIdx.x + off]; }
    __syncthreads();
  }
  float mean = rs[0] / DMODEL;
  float var  = rq[0] / DMODEL - mean * mean;
  float inv  = rsqrtf(var + 1e-5f);
  for (int i = threadIdx.x; i < DMODEL; i += 128)
    x[i] = (x[i] - mean) * inv * gg[i] + bb[i];
}

// ---------------- q = src + level_embed ----------------
__global__ void add_pos_kernel(const float* __restrict__ src,
                               const float* __restrict__ lev,
                               float* __restrict__ q, int total) {
  int i = blockIdx.x * blockDim.x + threadIdx.x;
  if (i >= total) return;
  int d  = i % DMODEL;
  int qi = (i / DMODEL) % LQ;
  q[i] = src[i] + lev[((qi >= HW) ? DMODEL : 0) + d];
}

// ---------------- deformable attention sampling ----------------
// one 64-thread block per (q, head, b); thread = output channel within head
__global__ __launch_bounds__(64)
void deform_sample_kernel(const float* __restrict__ value,
                          const float* __restrict__ off,
                          const float* __restrict__ aw,
                          float* __restrict__ attn) {
  int q = blockIdx.x;
  int h = blockIdx.y;
  int b = blockIdx.z;
  int d = threadIdx.x;

  // softmax over LEVELS*POINTS = 8 weights
  const float* awp = aw + ((long)b * LQ + q) * (NHEADS * NLEVELS * NPOINTS)
                        + h * (NLEVELS * NPOINTS);
  float a[8], mx = -1e30f;
#pragma unroll
  for (int i = 0; i < 8; ++i) { a[i] = awp[i]; mx = fmaxf(mx, a[i]); }
  float sum = 0.f;
#pragma unroll
  for (int i = 0; i < 8; ++i) { a[i] = __expf(a[i] - mx); sum += a[i]; }
  float rinv = 1.f / sum;

  int pq = q & (HW - 1);
  int px = pq & (WW - 1);
  int py = pq >> 5;      // / WW

  const float* offp = off + ((long)b * LQ + q) * (NHEADS * NLEVELS * NPOINTS * 2)
                          + h * (NLEVELS * NPOINTS * 2);
  float acc = 0.f;
#pragma unroll
  for (int lev = 0; lev < NLEVELS; ++lev) {
    const float* vb = value + ((long)b * LQ + (long)lev * HW) * DMODEL + h * HD + d;
#pragma unroll
    for (int p = 0; p < NPOINTS; ++p) {
      float ox = offp[(lev * NPOINTS + p) * 2 + 0];
      float oy = offp[(lev * NPOINTS + p) * 2 + 1];
      float x = (float)px + ox;
      float y = (float)py + oy;
      float x0f = floorf(x), y0f = floorf(y);
      int   x0 = (int)x0f,   y0 = (int)y0f;
      float wx1 = x - x0f, wy1 = y - y0f;
      float wx0 = 1.f - wx1, wy0 = 1.f - wy1;
      float wgt = a[lev * NPOINTS + p] * rinv;
      float s = 0.f;
      if (x0 >= 0 && x0 < WW && y0 >= 0 && y0 < HH)         s += wx0 * wy0 * vb[(long)(y0 * WW + x0) * DMODEL];
      if (x0+1 >= 0 && x0+1 < WW && y0 >= 0 && y0 < HH)     s += wx1 * wy0 * vb[(long)(y0 * WW + x0 + 1) * DMODEL];
      if (x0 >= 0 && x0 < WW && y0+1 >= 0 && y0+1 < HH)     s += wx0 * wy1 * vb[(long)((y0 + 1) * WW + x0) * DMODEL];
      if (x0+1 >= 0 && x0+1 < WW && y0+1 >= 0 && y0+1 < HH) s += wx1 * wy1 * vb[(long)((y0 + 1) * WW + x0 + 1) * DMODEL];
      acc += wgt * s;
    }
  }
  attn[((long)b * LQ + q) * DMODEL + h * HD + d] = acc;
}

// ---------------- out_tokens = src[:hw] + src[hw:] ----------------
__global__ void modality_sum_kernel(const float* __restrict__ src,
                                    float* __restrict__ osum, int total) {
  int i = blockIdx.x * blockDim.x + threadIdx.x;
  if (i >= total) return;
  int d = i % DMODEL;
  int p = (i / DMODEL) % HW;
  int b = i / (DMODEL * HW);
  const float* s = src + (long)b * LQ * DMODEL;
  osum[i] = s[(long)p * DMODEL + d] + s[(long)(p + HW) * DMODEL + d];
}

// ---------------- host-side orchestration ----------------
extern "C" void kernel_launch(void* const* d_in, const int* in_sizes, int n_in,
                              void* d_out, int out_size, void* d_ws, size_t ws_size,
                              hipStream_t stream) {
  const float* input_v    = (const float*)d_in[0];
  const float* input_i    = (const float*)d_in[1];
  const float* conv_in_w  = (const float*)d_in[2];
  const float* conv_in_b  = (const float*)d_in[3];
  const float* gn_in_g    = (const float*)d_in[4];
  const float* gn_in_b    = (const float*)d_in[5];
  const float* level_emb  = (const float*)d_in[6];
  const float* off_w      = (const float*)d_in[7];
  const float* off_b      = (const float*)d_in[8];
  const float* aw_w       = (const float*)d_in[9];
  const float* aw_b       = (const float*)d_in[10];
  const float* val_w      = (const float*)d_in[11];
  const float* val_b      = (const float*)d_in[12];
  const float* out_w      = (const float*)d_in[13];
  const float* out_b      = (const float*)d_in[14];
  const float* ln1_g      = (const float*)d_in[15];
  const float* ln1_b      = (const float*)d_in[16];
  const float* ln2_g      = (const float*)d_in[17];
  const float* ln2_b      = (const float*)d_in[18];
  const float* ff1_w      = (const float*)d_in[19];
  const float* ff1_b      = (const float*)d_in[20];
  const float* ff2_w      = (const float*)d_in[21];
  const float* ff2_b      = (const float*)d_in[22];
  const float* conv_out_w = (const float*)d_in[23];
  const float* conv_out_b = (const float*)d_in[24];
  const float* gn_out_g   = (const float*)d_in[25];
  const float* gn_out_b   = (const float*)d_in[26];
  float* dout = (float*)d_out;

  // workspace partition (~90 MB of fp32)
  float* ws   = (float*)d_ws;
  float* src  = ws;                                   // B*LQ*D
  float* qb   = src  + (long)BD * LQ * DMODEL;        // B*LQ*D (also attn buffer)
  float* valb = qb   + (long)BD * LQ * DMODEL;        // B*LQ*D (also osum buffer)
  float* hbuf = valb + (long)BD * LQ * DMODEL;        // B*LQ*FFN
  float* offb = hbuf + (long)BD * LQ * FFN_DIM;       // B*LQ*128
  float* awb  = offb + (long)BD * LQ * NHEADS * NLEVELS * NPOINTS * 2;  // B*LQ*64
  float* attn = qb;      // q consumed before sampling
  float* osum = valb;    // value consumed before final conv

  const long LQD = (long)LQ * DMODEL;

  // ---- conv_in (as GEMM into token layout) + groupnorm, per modality ----
  for (int mod = 0; mod < 2; ++mod) {
    const float* X = (mod == 0) ? input_v : input_i;
    // tok[p,d] = sum_c X[b,c,p]*W[d,c] + b[d]
    gemm_bf16_wmma<1, false, false, false, true>
        <<<dim3(DMODEL / TN, HW / TM, BD), 256, 0, stream>>>(
        X, (long)CCH * HW, 1, HW,
        conv_in_w, 0, 1, CCH,
        src + (long)mod * HW * DMODEL, LQD, DMODEL,
        conv_in_b, nullptr, 0, 0, CCH);
    group_norm_kernel<<<dim3(32, BD), 256, 0, stream>>>(
        src + (long)mod * HW * DMODEL, LQD, gn_in_g, gn_in_b,
        DMODEL / 32, HW, 1, DMODEL);
  }

  const int total = BD * LQ * DMODEL;

  for (int l = 0; l < NLAYERS; ++l) {
    // q = src + pos
    add_pos_kernel<<<(total + 255) / 256, 256, 0, stream>>>(src, level_emb, qb, total);

    // value = src @ val_w + val_b
    gemm_bf16_wmma<1, false, false, true, false>
        <<<dim3(DMODEL / TN, LQ / TM, BD), 256, 0, stream>>>(
        src, LQD, DMODEL, 1,
        val_w + (long)l * DMODEL * DMODEL, 0, DMODEL, 1,
        valb, LQD, DMODEL,
        val_b + (long)l * DMODEL, nullptr, 0, 0, DMODEL);

    // off = q @ off_w + off_b   (N = 128)
    gemm_bf16_wmma<1, false, false, true, false>
        <<<dim3(128 / TN, LQ / TM, BD), 256, 0, stream>>>(
        qb, LQD, DMODEL, 1,
        off_w + (long)l * DMODEL * 128, 0, 128, 1,
        offb, (long)LQ * 128, 128,
        off_b + (long)l * 128, nullptr, 0, 0, DMODEL);

    // aw = q @ aw_w + aw_b     (N = 64)
    gemm_bf16_wmma<1, false, false, true, false>
        <<<dim3(64 / TN, LQ / TM, BD), 256, 0, stream>>>(
        qb, LQD, DMODEL, 1,
        aw_w + (long)l * DMODEL * 64, 0, 64, 1,
        awb, (long)LQ * 64, 64,
        aw_b + (long)l * 64, nullptr, 0, 0, DMODEL);

    // deformable sampling (softmax fused) -> attn
    deform_sample_kernel<<<dim3(LQ, NHEADS, BD), 64, 0, stream>>>(valb, offb, awb, attn);

    // src = src + attn @ out_w + out_b
    gemm_bf16_wmma<1, false, true, true, false>
        <<<dim3(DMODEL / TN, LQ / TM, BD), 256, 0, stream>>>(
        attn, LQD, DMODEL, 1,
        out_w + (long)l * DMODEL * DMODEL, 0, DMODEL, 1,
        src, LQD, DMODEL,
        out_b + (long)l * DMODEL,
        src, LQD, DMODEL, DMODEL);

    layer_norm_kernel<<<dim3(LQ, BD), 128, 0, stream>>>(
        src, ln1_g + (long)l * 2 * DMODEL, ln1_b + (long)l * 2 * DMODEL);

    // h = relu(src @ ff1 + b1)
    gemm_bf16_wmma<1, true, false, true, false>
        <<<dim3(FFN_DIM / TN, LQ / TM, BD), 256, 0, stream>>>(
        src, LQD, DMODEL, 1,
        ff1_w + (long)l * DMODEL * FFN_DIM, 0, FFN_DIM, 1,
        hbuf, (long)LQ * FFN_DIM, FFN_DIM,
        ff1_b + (long)l * FFN_DIM, nullptr, 0, 0, DMODEL);

    // src = src + h @ ff2 + b2
    gemm_bf16_wmma<1, false, true, true, false>
        <<<dim3(DMODEL / TN, LQ / TM, BD), 256, 0, stream>>>(
        hbuf, (long)LQ * FFN_DIM, FFN_DIM, 1,
        ff2_w + (long)l * FFN_DIM * DMODEL, 0, DMODEL, 1,
        src, LQD, DMODEL,
        ff2_b + (long)l * DMODEL,
        src, LQD, DMODEL, FFN_DIM);

    layer_norm_kernel<<<dim3(LQ, BD), 128, 0, stream>>>(
        src, ln2_g + (long)l * 2 * DMODEL, ln2_b + (long)l * 2 * DMODEL);
  }

  // ---- osum = src[:hw] + src[hw:] ----
  const int totOut = BD * HW * DMODEL;
  modality_sum_kernel<<<(totOut + 255) / 256, 256, 0, stream>>>(src, osum, totOut);

  // ---- conv_out: y[c,p] = sum_d W[c,d]*osum[p,d] + b[c]  (bias per-M) ----
  gemm_bf16_wmma<2, false, false, true, true>
      <<<dim3(HW / TN, CCH / TM, BD), 256, 0, stream>>>(
      conv_out_w, 0, DMODEL, 1,
      osum, (long)HW * DMODEL, 1, DMODEL,
      dout, (long)CCH * HW, HW,
      conv_out_b, nullptr, 0, 0, DMODEL);

  // ---- final groupnorm on NCHW output (32 groups x 24 channels) ----
  group_norm_kernel<<<dim3(32, BD), 256, 0, stream>>>(
      dout, (long)CCH * HW, gn_out_g, gn_out_b,
      CCH / 32, HW, HW, 1);
}
